// DiGCL_1408749273635
// MI455X (gfx1250) — compile-verified, matrix operand order checked by reading, not run on hardware
//
#include <hip/hip_runtime.h>
#include <hip/hip_bf16.h>

typedef __attribute__((ext_vector_type(16))) _Float16 v16h;
typedef __attribute__((ext_vector_type(8)))  _Float16 v8h;
typedef __attribute__((ext_vector_type(8)))  float    v8f;

// ---------------------------------------------------------------------------
// Degree / normalization kernels
// ---------------------------------------------------------------------------
__global__ void deg_init_kernel(float* __restrict__ deg, int n) {
    int i = blockIdx.x * blockDim.x + threadIdx.x;
    if (i < n) deg[i] = 1.0f;                 // self-loop weight
}

__global__ void deg_edge_kernel(const int* __restrict__ col,
                                const float* __restrict__ w,
                                float* __restrict__ deg, int E) {
    int e = blockIdx.x * blockDim.x + threadIdx.x;
    if (e < E) atomicAdd(&deg[col[e]], w[e]);
}

__global__ void dinv_kernel(float* __restrict__ deg, int n) {
    int i = blockIdx.x * blockDim.x + threadIdx.x;
    if (i < n) {
        float d = deg[i];
        deg[i] = (d > 0.0f) ? rsqrtf(d) : 0.0f;   // in-place deg -> dinv
    }
}

// ---------------------------------------------------------------------------
// WMMA GEMM: out[M,NT] = A[M,128] @ W[128,NT]   (optionally A := relu(A + bias))
// Block: 256 threads = 8 waves; block tile = MT rows x NT cols.
// 8 waves = (MT/16) M-tiles x WPM waves each; each wave owns NPW N-tiles and
// reuses its A fragment NPW times per k-step.
// LDS rows padded to 136 halfs (272B = 68 dwords) -> fragment lanes hit
// disjoint banks while keeping 16B alignment for ds_load_b128.
// ---------------------------------------------------------------------------
template <int MT, int NT, bool FUSED>
__global__ void __launch_bounds__(256)
gemm_wmma_kernel(const float* __restrict__ A, const float* __restrict__ W,
                 const float* __restrict__ bias, float* __restrict__ out, int M) {
    constexpr int K  = 128;
    constexpr int KP = K + 8;                 // padded LDS row (halfs)
    __shared__ __attribute__((aligned(16))) _Float16 sA[MT * KP];
    __shared__ __attribute__((aligned(16))) _Float16 sB[NT * KP];

    const int tid = threadIdx.x;
    const int m0  = blockIdx.x * MT;
    const bool full = (m0 + MT <= M);

    // Stage W (K x NT, row-major) transposed into sB[n*KP + k] as f16.
    for (int i = tid; i < K * NT; i += 256) {
        int k = i / NT, n = i % NT;           // consecutive tid -> consecutive n: coalesced
        sB[n * KP + k] = (_Float16)W[i];
    }
    // Stage A tile (MT x K) into sA; fuse relu(A + bias) for layer 2.
    if (full) {
        for (int i = tid; i < MT * K; i += 256) {
            int r = i / K, k = i % K;         // consecutive tid -> consecutive k: coalesced
            float v = A[(size_t)(m0 + r) * K + k];
            if (FUSED) { v += bias[k]; v = fmaxf(v, 0.0f); }
            sA[r * KP + k] = (_Float16)v;
        }
    } else {
        for (int i = tid; i < MT * K; i += 256) {
            int r = i / K, k = i % K;
            int gr = m0 + r;
            float v = (gr < M) ? A[(size_t)gr * K + k] : 0.0f;
            if (FUSED) { v += bias[k]; v = fmaxf(v, 0.0f); }
            sA[r * KP + k] = (_Float16)v;
        }
    }
    __syncthreads();

    const int wave = tid >> 5;
    const int lane = tid & 31;
    constexpr int TM  = MT / 16;              // M tiles per block
    constexpr int WPM = 8 / TM;               // waves sharing one M tile
    constexpr int NPW = (NT / 16) / WPM;      // N tiles per wave
    const int wm   = wave / WPM;
    const int wn0  = (wave % WPM) * NPW;      // first N tile of this wave
    const int lrow = lane & 15;
    const int half = lane >> 4;

    // A fragment: lane -> M row; element j -> K = k0 + half*8 + (j&7) + (j>=8)*16
    const _Float16* aBase = sA + (wm * 16 + lrow) * KP + half * 8;
    // B fragment: lane -> N col; same K packing.
    const _Float16* bRow  = sB + lrow * KP + half * 8;

    v8f acc[NPW] = {};
#pragma unroll
    for (int k0 = 0; k0 < K; k0 += 32) {
        v8h alo = *(const v8h*)(aBase + k0);
        v8h ahi = *(const v8h*)(aBase + k0 + 16);
        v16h a;
#pragma unroll
        for (int j = 0; j < 8; ++j) { a[j] = alo[j]; a[j + 8] = ahi[j]; }
#pragma unroll
        for (int t = 0; t < NPW; ++t) {
            const _Float16* bBase = bRow + (wn0 + t) * 16 * KP;
            v8h blo = *(const v8h*)(bBase + k0);
            v8h bhi = *(const v8h*)(bBase + k0 + 16);
            v16h b;
#pragma unroll
            for (int j = 0; j < 8; ++j) { b[j] = blo[j]; b[j + 8] = bhi[j]; }
            acc[t] = __builtin_amdgcn_wmma_f32_16x16x32_f16(
                false, a, false, b, (short)0, acc[t], false, false);
        }
    }

    // C/D layout: lane(0-15)->N, VGPR v -> M = v (+8 for upper-half lanes).
    const int mbase = m0 + wm * 16 + half * 8;
    if (full) {
#pragma unroll
        for (int t = 0; t < NPW; ++t) {
            float* p = out + (size_t)mbase * NT + (wn0 + t) * 16 + lrow;
#pragma unroll
            for (int v = 0; v < 8; ++v) p[(size_t)v * NT] = acc[t][v];
        }
    } else {
#pragma unroll
        for (int t = 0; t < NPW; ++t) {
            const int ncol = (wn0 + t) * 16 + lrow;
#pragma unroll
            for (int v = 0; v < 8; ++v) {
                int mr = mbase + v;
                if (mr < M) out[(size_t)mr * NT + ncol] = acc[t][v];
            }
        }
    }
}

// ---------------------------------------------------------------------------
// Aggregation kernels
// ---------------------------------------------------------------------------
// agg[i] = dinv[i]^2 * h[i]   (self-loop term; also initializes the buffer)
template <int CH>
__global__ void self_loop_kernel(const float* __restrict__ h,
                                 const float* __restrict__ dinv,
                                 float* __restrict__ agg, int n) {
    constexpr int C4 = CH / 4;
    int idx = blockIdx.x * blockDim.x + threadIdx.x;
    if (idx >= n * C4) return;
    int i = idx / C4, c4 = idx % C4;
    float s = dinv[i]; s = s * s;
    float4 hv = ((const float4*)h)[(size_t)i * C4 + c4];
    float4 o  = { s * hv.x, s * hv.y, s * hv.z, s * hv.w };
    ((float4*)agg)[(size_t)i * C4 + c4] = o;
}

// agg[col[e]] += dinv[row[e]] * w[e] * dinv[col[e]] * h[row[e]]  (4 ch / thread)
template <int CH>
__global__ void agg_edge_kernel(const float* __restrict__ h,
                                const int* __restrict__ row,
                                const int* __restrict__ col,
                                const float* __restrict__ w,
                                const float* __restrict__ dinv,
                                float* __restrict__ agg, long long total) {
    constexpr int C4 = CH / 4;
    long long idx = (long long)blockIdx.x * blockDim.x + threadIdx.x;
    if (idx >= total) return;
    int e  = (int)(idx / C4);
    int c4 = (int)(idx % C4);
    int r = row[e], c = col[e];
    float s = dinv[r] * w[e] * dinv[c];
    float4 hv = ((const float4*)h)[(size_t)r * C4 + c4];
    float* dst = agg + ((size_t)c * CH + c4 * 4);
    atomicAdd(dst + 0, s * hv.x);
    atomicAdd(dst + 1, s * hv.y);
    atomicAdd(dst + 2, s * hv.z);
    atomicAdd(dst + 3, s * hv.w);
}

template <int CH>
__global__ void bias_relu_kernel(float* __restrict__ out,
                                 const float* __restrict__ b, int n) {
    int idx = blockIdx.x * blockDim.x + threadIdx.x;
    if (idx >= n * CH) return;
    float v = out[idx] + b[idx % CH];
    out[idx] = fmaxf(v, 0.0f);
}

// ---------------------------------------------------------------------------
// Launch
// ---------------------------------------------------------------------------
extern "C" void kernel_launch(void* const* d_in, const int* in_sizes, int n_in,
                              void* d_out, int out_size, void* d_ws, size_t ws_size,
                              hipStream_t stream) {
    const float* x  = (const float*)d_in[0];   // [N,128]
    const int*   ei = (const int*)d_in[1];     // [2,E]
    const float* ew = (const float*)d_in[2];   // [E]
    const float* W1 = (const float*)d_in[3];   // [128,128]
    const float* b1 = (const float*)d_in[4];   // [128]
    const float* W2 = (const float*)d_in[5];   // [128,64]
    const float* b2 = (const float*)d_in[6];   // [64]
    float* out = (float*)d_out;                // [N,64]

    const int N = in_sizes[0] / 128;
    const int E = in_sizes[2];
    const int* row = ei;
    const int* col = ei + E;

    // Workspace layout: dinv [N f32] | h1 [N*128 f32] | agg1 [N*128 f32]
    char*  ws   = (char*)d_ws;
    float* dinv = (float*)ws;
    size_t off  = ((size_t)N * 4 + 4095) & ~(size_t)4095;
    float* h1   = (float*)(ws + off);
    float* agg1 = (float*)(ws + off + (size_t)N * 128 * 4);
    float* h2   = h1;  // h1 dead once GEMM2 consumes agg1

    dim3 blk(256);
    auto nb = [](long long work) { return (unsigned)((work + 255) / 256); };

    // Degrees -> dinv
    deg_init_kernel<<<nb(N), blk, 0, stream>>>(dinv, N);
    deg_edge_kernel<<<nb(E), blk, 0, stream>>>(col, ew, dinv, E);
    dinv_kernel<<<nb(N), blk, 0, stream>>>(dinv, N);

    // Layer 1: h1 = x @ W1 ; agg1 = norm-scatter(h1)
    gemm_wmma_kernel<64, 128, false>
        <<<(N + 63) / 64, blk, 0, stream>>>(x, W1, nullptr, h1, N);
    self_loop_kernel<128><<<nb((long long)N * 32), blk, 0, stream>>>(h1, dinv, agg1, N);
    long long tot1 = (long long)E * 32;
    agg_edge_kernel<128><<<nb(tot1), blk, 0, stream>>>(h1, row, col, ew, dinv, agg1, tot1);

    // Layer 2: h2 = relu(agg1 + b1) @ W2 (fused into A load); scatter into d_out
    gemm_wmma_kernel<64, 64, true>
        <<<(N + 63) / 64, blk, 0, stream>>>(agg1, W2, b1, h2, N);
    self_loop_kernel<64><<<nb((long long)N * 16), blk, 0, stream>>>(h2, dinv, out, N);
    long long tot2 = (long long)E * 16;
    agg_edge_kernel<64><<<nb(tot2), blk, 0, stream>>>(h2, row, col, ew, dinv, out, tot2);
    bias_relu_kernel<64><<<nb((long long)N * 64), blk, 0, stream>>>(out, b2, N);
}
